// StaticGraphLSTMCell__36386962932006
// MI455X (gfx1250) — matrix operand order repaired and checked by reading, compiled, not run on hardware
//
#include <hip/hip_runtime.h>
#include <hip/hip_bf16.h>
#include <stdint.h>

// ---------------------------------------------------------------------------
// StaticGraphLSTMCell for MI455X (gfx1250, wave32, WMMA bf16, async-to-LDS)
// B=512, N=21, I=256, H=1024, NT=5, K = I+H = 1280, 4H = 4096
// ---------------------------------------------------------------------------

#define BDIM   512
#define NNODE  21
#define IDIM   256
#define HDIM   1024
#define NTYPE  5
#define KDIM   1280
#define FOURH  4096

typedef __attribute__((ext_vector_type(16))) __bf16 bf16x16;
typedef __attribute__((ext_vector_type(8)))  float  v8f;

// f32 -> bf16 bits, round-to-nearest-even
__device__ __forceinline__ unsigned short f2bf(float f) {
    unsigned u = __float_as_uint(f);
    unsigned r = (u + 0x7FFFu + ((u >> 16) & 1u)) >> 16;
    return (unsigned short)r;
}

// Async DMA: copy 16B per lane from global to LDS (tracked by ASYNCcnt),
// plus a second 16B chunk at +16 on both addresses.
__device__ __forceinline__ void async_copy32(unsigned ldsOff, const void* gptr) {
    asm volatile("global_load_async_to_lds_b128 %0, %1, off"
                 :: "v"(ldsOff), "v"(gptr) : "memory");
    asm volatile("global_load_async_to_lds_b128 %0, %1, off offset:16"
                 :: "v"(ldsOff), "v"(gptr) : "memory");
}

// ---------------------------------------------------------------------------
// Kernel 1: gx = L1-row-normalize(G); gx_out = L1-row-normalize(gx)
// ---------------------------------------------------------------------------
__global__ void gx_kernel(const float* __restrict__ G,
                          float* __restrict__ gxw,     // ws, used by mix kernel
                          float* __restrict__ gxout) { // third output
    int m = threadIdx.x;
    if (m < NNODE) {
        float s = 0.f;
        for (int n = 0; n < NNODE; ++n) s += fabsf(G[m * NNODE + n]);
        s = fmaxf(s, 1e-12f);
        float s2 = 0.f;
        for (int n = 0; n < NNODE; ++n) {
            float v = G[m * NNODE + n] / s;
            gxw[m * NNODE + n] = v;
            s2 += fabsf(v);
        }
        s2 = fmaxf(s2, 1e-12f);
        for (int n = 0; n < NNODE; ++n)
            gxout[m * NNODE + n] = gxw[m * NNODE + n] / s2;
    }
}

// ---------------------------------------------------------------------------
// Kernel 2a: pack X[n][b][k] = bf16( k<256 ? input[b][n][k] : hx[b][n][k-256] )
// ---------------------------------------------------------------------------
__global__ void convert_x_kernel(const float* __restrict__ input,
                                 const float* __restrict__ hx,
                                 unsigned short* __restrict__ xbf) {
    int idx = blockIdx.x * blockDim.x + threadIdx.x;    // [21][512][1280]
    int k = idx % KDIM;
    int r = idx / KDIM;
    int b = r % BDIM;
    int n = r / BDIM;
    float v = (k < IDIM) ? input[((size_t)b * NNODE + n) * IDIM + k]
                         : hx[((size_t)b * NNODE + n) * HDIM + (k - IDIM)];
    xbf[idx] = f2bf(v);
}

// ---------------------------------------------------------------------------
// Kernel 2b: pack W[t][o][k] = bf16( k<256 ? w_ih[t][o][k] : w_hh[t][o][k-256] )
// ---------------------------------------------------------------------------
__global__ void convert_w_kernel(const float* __restrict__ w_ih,
                                 const float* __restrict__ w_hh,
                                 unsigned short* __restrict__ wbf) {
    int idx = blockIdx.x * blockDim.x + threadIdx.x;    // [5][4096][1280]
    int k = idx % KDIM;
    int r = idx / KDIM;       // t*4096 + o
    float v = (k < IDIM) ? w_ih[(size_t)r * IDIM + k]
                         : w_hh[(size_t)r * HDIM + (k - IDIM)];
    wbf[idx] = f2bf(v);
}

// ---------------------------------------------------------------------------
// Kernel 3: per-node GEMM, bf16 WMMA, f32 accumulate, async double buffering.
//   gates_pre[bl][node][o] = sum_k X[node][b0+b][k] * W[type][o][k] + bias[type][o]
// Block tile 128(M) x 128(N) x 32(K); 256 threads = 8 waves, wave tile 64x32.
// ---------------------------------------------------------------------------
__global__ __launch_bounds__(256)
void gemm_kernel(const unsigned short* __restrict__ Xbf,   // [21][512][1280]
                 const unsigned short* __restrict__ Wbf,   // [5][4096][1280]
                 const float* __restrict__ bias,           // [5][4096]
                 const int* __restrict__ node_types,
                 float* __restrict__ gates,                // [Bc][21][4096]
                 int b0) {
    // LDS rows padded to 40 ushorts (80B = 5 x uint4) to avoid bank conflicts.
    // Ping-pong double buffer, filled by global_load_async_to_lds_b128.
    __shared__ uint4 smemA[2][128 * 5];
    __shared__ uint4 smemB[2][128 * 5];

    const int ntile = blockIdx.x;            // 0..31  (o tiles of 128)
    const int mtile = blockIdx.y;            // chunk-local M tiles of 128
    const int node  = blockIdx.z;            // 0..20
    const int nt    = node_types[node];

    const int tid    = threadIdx.x;
    const int lane   = tid & 31;
    const int wid    = tid >> 5;
    const int waveM  = wid >> 2;             // 0..1
    const int waveN  = wid & 3;              // 0..3
    const int laneLo = lane & 15;
    const int hi     = lane >> 4;

    // global->LDS copy mapping: each thread moves 32B of A and 32B of B
    const int rowA  = tid >> 1;              // 0..127
    const int halfA = tid & 1;

    const unsigned short* aRow =
        Xbf + ((size_t)node * BDIM + (size_t)(b0 + mtile * 128 + rowA)) * KDIM;
    const unsigned short* bRow =
        Wbf + ((size_t)nt * FOURH + (size_t)(ntile * 128 + rowA)) * KDIM;

    const unsigned ldsA0 = (unsigned)(uintptr_t)&smemA[0][rowA * 5 + halfA * 2];
    const unsigned ldsA1 = (unsigned)(uintptr_t)&smemA[1][rowA * 5 + halfA * 2];
    const unsigned ldsB0 = (unsigned)(uintptr_t)&smemB[0][rowA * 5 + halfA * 2];
    const unsigned ldsB1 = (unsigned)(uintptr_t)&smemB[1][rowA * 5 + halfA * 2];

    v8f acc[4][2] = {};

    // Prologue: request K-tile 0 into buffer 0 (4 async ops per wave).
    async_copy32(ldsA0, aRow + 0 + halfA * 16);
    async_copy32(ldsB0, bRow + 0 + halfA * 16);

    int cur = 0;
    for (int k0 = 0; k0 < KDIM; k0 += 32) {
        if (k0 + 32 < KDIM) {
            // Request next K-tile into the other buffer while computing.
            int kn = k0 + 32;
            async_copy32(cur ? ldsA0 : ldsA1, aRow + kn + halfA * 16);
            async_copy32(cur ? ldsB0 : ldsB1, bRow + kn + halfA * 16);
            // Only the 4 older async ops (current tile) must have landed.
            asm volatile("s_wait_asynccnt 0x4" ::: "memory");
        } else {
            asm volatile("s_wait_asynccnt 0x0" ::: "memory");
        }
        __syncthreads();

        const uint4* sA = smemA[cur];
        const uint4* sB = smemB[cur];

        // B fragments: lane holds col o = laneLo, elems 0..7 -> K = hi*8 + 0..7 ; elems 8..15 -> K = 16 + hi*8 + 0..7
        union { bf16x16 v; uint4 q[2]; } bfr[2];
#pragma unroll
        for (int nj = 0; nj < 2; ++nj) {
            int ro = waveN * 32 + nj * 16 + laneLo;
            bfr[nj].q[0] = sB[ro * 5 + hi];
            bfr[nj].q[1] = sB[ro * 5 + 2 + hi];
        }
#pragma unroll
        for (int mi = 0; mi < 4; ++mi) {
            // A fragment: lane holds row m = laneLo,
            // elems 0..7 -> K = hi*8 + 0..7 ; elems 8..15 -> K = 16 + hi*8 + 0..7
            int rm = waveM * 64 + mi * 16 + laneLo;
            union { bf16x16 v; uint4 q[2]; } af;
            af.q[0] = sA[rm * 5 + hi];        // bytes rm*80 + hi*16
            af.q[1] = sA[rm * 5 + 2 + hi];    // bytes rm*80 + 32 + hi*16
#pragma unroll
            for (int nj = 0; nj < 2; ++nj) {
                acc[mi][nj] = __builtin_amdgcn_wmma_f32_16x16x32_bf16(
                    false, af.v, false, bfr[nj].v, (short)0, acc[mi][nj],
                    false, false);
            }
        }
        __syncthreads();   // all waves done reading buf[cur] before it is refilled
        cur ^= 1;
    }

    // Epilogue: add bias, store f32 gates_pre (chunk-local batch index)
    const int bl0 = mtile * 128;
#pragma unroll
    for (int nj = 0; nj < 2; ++nj) {
        int o = ntile * 128 + waveN * 32 + nj * 16 + laneLo;
        float bv = bias[(size_t)nt * FOURH + o];
#pragma unroll
        for (int mi = 0; mi < 4; ++mi) {
#pragma unroll
            for (int v = 0; v < 8; ++v) {
                int m = waveM * 64 + mi * 16 + hi * 8 + v;
                size_t idx = ((size_t)(bl0 + m) * NNODE + node) * FOURH + o;
                gates[idx] = acc[mi][nj][v] + bv;
            }
        }
    }
}

// ---------------------------------------------------------------------------
// Kernel 4: graph mix over nodes + clockwork-LSTM activations.
// One thread per (b, j); computes all 21 output nodes m.
// ---------------------------------------------------------------------------
__global__ __launch_bounds__(256)
void mix_act_kernel(const float* __restrict__ gates,  // [Bc][21][4096]
                    const float* __restrict__ gxw,    // [21][21]
                    const float* __restrict__ cx,     // [512][21][1024]
                    const int* __restrict__ tptr,
                    float* __restrict__ hy,           // [512][21][1024]
                    float* __restrict__ cyo,          // [512][21][1024]
                    int b0) {
    __shared__ float gxs[NNODE * NNODE];
    int tid = threadIdx.x;
    for (int i = tid; i < NNODE * NNODE; i += 256) gxs[i] = gxw[i];
    __syncthreads();

    int bl = blockIdx.x >> 2;                 // chunk-local batch
    int j  = ((blockIdx.x & 3) << 8) + tid;   // hidden index 0..1023
    int b  = b0 + bl;

    // clockwork mask: phase = floor(j/1023*8 + 1) in {1..9}; t+1 mod phase < 0.01
    float t1 = (float)(*tptr) + 1.0f;
    float ph = floorf((float)j * (8.0f / 1023.0f) + 1.0f);
    float cm = (fmodf(t1, ph) < 0.01f) ? 1.0f : 0.0f;

    float gp[NNODE][4];
    const float* gbase = gates + (size_t)bl * NNODE * FOURH + j;
#pragma unroll
    for (int n = 0; n < NNODE; ++n) {
        const float* gn = gbase + (size_t)n * FOURH;
        gp[n][0] = gn[0];
        gp[n][1] = gn[HDIM];
        gp[n][2] = gn[2 * HDIM];
        gp[n][3] = gn[3 * HDIM];
    }

    for (int m = 0; m < NNODE; ++m) {
        float s0 = 0.f, s1 = 0.f, s2 = 0.f, s3 = 0.f;
#pragma unroll
        for (int n = 0; n < NNODE; ++n) {
            float w = gxs[m * NNODE + n];
            s0 = fmaf(w, gp[n][0], s0);
            s1 = fmaf(w, gp[n][1], s1);
            s2 = fmaf(w, gp[n][2], s2);
            s3 = fmaf(w, gp[n][3], s3);
        }
        float ig = 1.0f / (1.0f + __expf(-s0));
        float fg = 1.0f / (1.0f + __expf(-s1));
        float gg = tanhf(s2);
        float og = 1.0f / (1.0f + __expf(-s3));

        size_t oidx = ((size_t)b * NNODE + m) * HDIM + j;
        float c  = cx[oidx];
        float cy = cm * (fg * c + ig * gg) + (1.0f - cm) * c;
        cyo[oidx] = cy;
        hy[oidx]  = og * tanhf(cy);
    }
}

// ---------------------------------------------------------------------------
// Host launcher
// ---------------------------------------------------------------------------
extern "C" void kernel_launch(void* const* d_in, const int* in_sizes, int n_in,
                              void* d_out, int out_size, void* d_ws, size_t ws_size,
                              hipStream_t stream) {
    const float* input      = (const float*)d_in[0];
    const float* hx         = (const float*)d_in[1];
    const float* cx         = (const float*)d_in[2];
    const float* G          = (const float*)d_in[3];
    const float* w_ih       = (const float*)d_in[4];
    const float* w_hh       = (const float*)d_in[5];
    const float* b_hh       = (const float*)d_in[6];
    const int*   node_types = (const int*)d_in[7];
    const int*   tptr       = (const int*)d_in[8];

    float* hy  = (float*)d_out;
    float* cyo = hy + (size_t)BDIM * NNODE * HDIM;
    float* gxo = cyo + (size_t)BDIM * NNODE * HDIM;

    char* ws = (char*)d_ws;
    const size_t gxOff  = 0;
    const size_t xOff   = 2048;
    const size_t xBytes = (size_t)NNODE * BDIM * KDIM * sizeof(unsigned short);
    const size_t wOff   = xOff + xBytes;
    const size_t wBytes = (size_t)NTYPE * FOURH * KDIM * sizeof(unsigned short);
    size_t gOff = (wOff + wBytes + 255) & ~(size_t)255;

    float*          gxw = (float*)(ws + gxOff);
    unsigned short* Xbf = (unsigned short*)(ws + xOff);
    unsigned short* Wbf = (unsigned short*)(ws + wOff);
    float*          gpr = (float*)(ws + gOff);

    // largest batch chunk whose gates_pre scratch fits in ws
    int Bc = 128;
    if (gOff + (size_t)512 * NNODE * FOURH * 4 <= ws_size)      Bc = 512;
    else if (gOff + (size_t)256 * NNODE * FOURH * 4 <= ws_size) Bc = 256;

    gx_kernel<<<1, 32, 0, stream>>>(G, gxw, gxo);

    {
        int tot = NNODE * BDIM * KDIM;          // 13,762,560 (divisible by 256)
        convert_x_kernel<<<tot / 256, 256, 0, stream>>>(input, hx, Xbf);
    }
    {
        int tot = NTYPE * FOURH * KDIM;         // 26,214,400 (divisible by 256)
        convert_w_kernel<<<tot / 256, 256, 0, stream>>>(w_ih, w_hh, Wbf);
    }

    for (int b0 = 0; b0 < BDIM; b0 += Bc) {
        dim3 grid(FOURH / 128, Bc / 128, NNODE);
        gemm_kernel<<<grid, 256, 0, stream>>>(Xbf, Wbf, b_hh, node_types, gpr, b0);
        mix_act_kernel<<<Bc * 4, 256, 0, stream>>>(gpr, gxw, cx, tptr, hy, cyo, b0);
    }
}
